// LinearGEGLU_64372969832561
// MI455X (gfx1250) — 8-bit matrix operand order repaired and checked by reading, compiled, not run on hardware
//
#include <hip/hip_runtime.h>
#include <math.h>

typedef __attribute__((ext_vector_type(8))) int v8i;

#define DIM_M   8192
#define DIM_K   2048
#define DIM_NO  8192
#define DIM_N2  16384
#define W_ELEMS (33554432)
#define X_ELEMS (16777216)
#define EPSQ 1e-5f

__global__ __launch_bounds__(256) void w_abssum(const float4* __restrict__ w4,
                                                float* __restrict__ partials) {
    __shared__ float red[256];
    float s = 0.f;
    const int n4 = W_ELEMS / 4;
    for (int i = blockIdx.x * 256 + threadIdx.x; i < n4; i += 1024 * 256) {
        float4 v = w4[i];
        s += fabsf(v.x) + fabsf(v.y) + fabsf(v.z) + fabsf(v.w);
    }
    red[threadIdx.x] = s;
    __syncthreads();
    for (int t = 128; t > 0; t >>= 1) {
        if (threadIdx.x < t) red[threadIdx.x] += red[threadIdx.x + t];
        __syncthreads();
    }
    if (threadIdx.x == 0) partials[blockIdx.x] = red[0];
}

__global__ __launch_bounds__(256) void w_finalize(const float* __restrict__ partials,
                                                  float* __restrict__ wsc) {
    __shared__ float red[256];
    float s = partials[threadIdx.x] + partials[threadIdx.x + 256] +
              partials[threadIdx.x + 512] + partials[threadIdx.x + 768];
    red[threadIdx.x] = s;
    __syncthreads();
    for (int t = 128; t > 0; t >>= 1) {
        if (threadIdx.x < t) red[threadIdx.x] += red[threadIdx.x + t];
        __syncthreads();
    }
    if (threadIdx.x == 0) {
        float mean = red[0] / (float)W_ELEMS;
        float wdeq = fmaxf(mean, EPSQ);
        wsc[0] = wdeq;
        wsc[1] = 1.0f / wdeq;
    }
}

__global__ __launch_bounds__(256) void w_quant(const float4* __restrict__ w4,
                                               const float* __restrict__ wsc,
                                               char4* __restrict__ wq4) {
    const float sc = wsc[1];
    const int i = blockIdx.x * 256 + threadIdx.x;
    float4 v = w4[i];
    char4 q;
    q.x = (signed char)(int)fminf(fmaxf(rintf(v.x * sc), -1.f), 1.f);
    q.y = (signed char)(int)fminf(fmaxf(rintf(v.y * sc), -1.f), 1.f);
    q.z = (signed char)(int)fminf(fmaxf(rintf(v.z * sc), -1.f), 1.f);
    q.w = (signed char)(int)fminf(fmaxf(rintf(v.w * sc), -1.f), 1.f);
    wq4[i] = q;
}

__global__ __launch_bounds__(256) void x_quant(const float* __restrict__ x,
                                               signed char* __restrict__ xq,
                                               float* __restrict__ adeq) {
    __shared__ float red[256];
    __shared__ float sA;
    const int row = blockIdx.x;
    const int tid = threadIdx.x;
    const float4* xr = (const float4*)(x + (size_t)row * DIM_K);
    float4 v0 = xr[tid * 2];
    float4 v1 = xr[tid * 2 + 1];
    float mx = fmaxf(fmaxf(fabsf(v0.x), fabsf(v0.y)), fmaxf(fabsf(v0.z), fabsf(v0.w)));
    mx = fmaxf(mx, fmaxf(fmaxf(fabsf(v1.x), fabsf(v1.y)), fmaxf(fabsf(v1.z), fabsf(v1.w))));
    red[tid] = mx;
    __syncthreads();
    for (int t = 128; t > 0; t >>= 1) {
        if (tid < t) red[tid] = fmaxf(red[tid], red[tid + t]);
        __syncthreads();
    }
    if (tid == 0) {
        float ma = fmaxf(red[0], EPSQ);
        sA = 127.0f / ma;
        adeq[row] = ma * (1.0f / 127.0f);
    }
    __syncthreads();
    const float a = sA;
#define Q8(v) (((int)fminf(fmaxf(rintf((v) * a), -128.f), 127.f)) & 0xFF)
    int lo = Q8(v0.x) | (Q8(v0.y) << 8) | (Q8(v0.z) << 16) | (Q8(v0.w) << 24);
    int hi = Q8(v1.x) | (Q8(v1.y) << 8) | (Q8(v1.z) << 16) | (Q8(v1.w) << 24);
#undef Q8
    ((int2*)(xq + (size_t)row * DIM_K))[tid] = make_int2(lo, hi);
}

__device__ __forceinline__ v8i lds_frag_a(const signed char* As, int rowLocal, int lane) {
    const int koff = (lane & 16) >> 1;
    const signed char* p = As + rowLocal * 64 + koff;
    int2 q0 = *(const int2*)(p + 0);
    int2 q1 = *(const int2*)(p + 16);
    int2 q2 = *(const int2*)(p + 32);
    int2 q3 = *(const int2*)(p + 48);
    v8i f;
    f[0] = q0.x; f[1] = q0.y; f[2] = q1.x; f[3] = q1.y;
    f[4] = q2.x; f[5] = q2.y; f[6] = q3.x; f[7] = q3.y;
    return f;
}

__device__ __forceinline__ v8i lds_frag_b(const signed char* Bs, int rowLocal, int lane) {
    const int khalf = (lane & 16) >> 1;
    const signed char* p = Bs + rowLocal * 64 + khalf;
    int2 q0 = *(const int2*)(p + 0);
    int2 q1 = *(const int2*)(p + 16);
    int2 q2 = *(const int2*)(p + 32);
    int2 q3 = *(const int2*)(p + 48);
    v8i f;
    f[0] = q0.x; f[1] = q0.y; f[2] = q1.x; f[3] = q1.y;
    f[4] = q2.x; f[5] = q2.y; f[6] = q3.x; f[7] = q3.y;
    return f;
}

__global__ __launch_bounds__(256) void geglu_gemm(
    const signed char* __restrict__ Xq,
    const signed char* __restrict__ Wq,
    const float* __restrict__ adeq,
    const float* __restrict__ wsc,
    const float* __restrict__ bias,
    float* __restrict__ out) {
    __shared__ __align__(16) signed char As[128 * 64];
    __shared__ __align__(16) signed char Bs[128 * 64];

    const int tid  = threadIdx.x;
    const int lane = tid & 31;
    const int wave = tid >> 5;
    const int wm   = wave & 3;
    const int wn   = wave >> 2;
    const int m_block = blockIdx.y * 128;
    const int n_block = blockIdx.x * 64;

    v8i acc[2][2][2] = {};

    for (int ks = 0; ks < DIM_K / 64; ++ks) {
        const int k0 = ks * 64;
        __syncthreads();
#pragma unroll
        for (int i = 0; i < 2; ++i) {
            const int c   = tid + i * 256;
            const int row = c >> 2;
            const int off = (c & 3) * 16;
            *(int4*)(As + row * 64 + off) =
                *(const int4*)(Xq + (size_t)(m_block + row) * DIM_K + k0 + off);
            const int ngl = (row < 64) ? (n_block + row)
                                       : (DIM_NO + n_block + row - 64);
            *(int4*)(Bs + row * 64 + off) =
                *(const int4*)(Wq + (size_t)ngl * DIM_K + k0 + off);
        }
        __syncthreads();

        const v8i a0 = lds_frag_a(As, wm * 32 + (lane & 15), lane);
        const v8i a1 = lds_frag_a(As, wm * 32 + 16 + (lane & 15), lane);
        v8i bf[2][2];
#pragma unroll
        for (int h = 0; h < 2; ++h)
#pragma unroll
            for (int ni = 0; ni < 2; ++ni)
                bf[h][ni] = lds_frag_b(Bs, h * 64 + wn * 32 + ni * 16 + (lane & 15), lane);

#pragma unroll
        for (int ni = 0; ni < 2; ++ni)
#pragma unroll
            for (int h = 0; h < 2; ++h) {
                acc[0][ni][h] = __builtin_amdgcn_wmma_i32_16x16x64_iu8(
                    true, a0, true, bf[h][ni], acc[0][ni][h], false, false);
                acc[1][ni][h] = __builtin_amdgcn_wmma_i32_16x16x64_iu8(
                    true, a1, true, bf[h][ni], acc[1][ni][h], false, false);
            }
    }

    const float wdeq = wsc[0];
    const int mrow_off = (lane & 16) >> 1;
    float sm[2][8];
#pragma unroll
    for (int mi = 0; mi < 2; ++mi)
#pragma unroll
        for (int r = 0; r < 8; ++r)
            sm[mi][r] = adeq[m_block + wm * 32 + mi * 16 + r + mrow_off] * wdeq;

#pragma unroll
    for (int mi = 0; mi < 2; ++mi) {
#pragma unroll
        for (int ni = 0; ni < 2; ++ni) {
            const int n = n_block + wn * 32 + ni * 16 + (lane & 15);
            const float bo = bias[n];
            const float bg = bias[n + DIM_NO];
#pragma unroll
            for (int r = 0; r < 8; ++r) {
                const int m = m_block + wm * 32 + mi * 16 + r + mrow_off;
                const float o = (float)acc[mi][ni][0][r] * sm[mi][r] + bo;
                const float g = (float)acc[mi][ni][1][r] * sm[mi][r] + bg;
                const float gl = 0.5f * g * (1.0f + erff(g * 0.70710678118654752f));
                out[(size_t)m * DIM_NO + n] = o * gl;
            }
        }
    }
}

extern "C" void kernel_launch(void* const* d_in, const int* in_sizes, int n_in,
                              void* d_out, int out_size, void* d_ws, size_t ws_size,
                              hipStream_t stream) {
    const float* x    = (const float*)d_in[0];
    const float* w    = (const float*)d_in[1];
    const float* bias = (const float*)d_in[2];
    float* out        = (float*)d_out;

    char* ws = (char*)d_ws;
    float* partials   = (float*)ws;
    float* wsc        = (float*)(ws + 4096);
    float* adeq       = (float*)(ws + 8192);
    signed char* Wq   = (signed char*)(ws + 65536);
    signed char* Xq   = (signed char*)(ws + 65536 + (size_t)W_ELEMS);

    w_abssum  <<<1024, 256, 0, stream>>>((const float4*)w, partials);
    w_finalize<<<1,    256, 0, stream>>>(partials, wsc);
    w_quant   <<<W_ELEMS / 4 / 256, 256, 0, stream>>>((const float4*)w, wsc, (char4*)Wq);
    x_quant   <<<DIM_M, 256, 0, stream>>>(x, Xq, adeq);

    dim3 grid(DIM_NO / 64, DIM_M / 128);
    geglu_gemm<<<grid, 256, 0, stream>>>(Xq, Wq, adeq, wsc, bias, out);
}
